// MSTACell_3083786518696
// MI455X (gfx1250) — matrix operand order repaired and checked by reading, not compiled, not run on hardware
//
#include <hip/hip_runtime.h>
#include <hip/hip_bf16.h>

// ---------------------------------------------------------------------------
// MSTACell fused implementation for gfx1250 (CDNA5, wave32, WMMA 16x16x32 f16)
//
// Shapes: B=16, T=12, N=2048, DI=32, DO=64, DE=16, K=2, heads=4, hd=16
// Heavy math: per gate, per (b,t): softmax(emb emb^T) @ x  (2048x2048 @ 2048x96)
// -> fused flash-style, f16 WMMA with f32 accumulation, async-LDS double buffer.
// ---------------------------------------------------------------------------

typedef __attribute__((ext_vector_type(16))) _Float16 v16h;
typedef __attribute__((ext_vector_type(8)))  float    v8f;

#define NN   2048      // nodes
#define BT   32        // B*K = 16*2
#define CIN  96        // DI + DO
#define TT   12        // T
#define KI   192       // 2*CIN (stacked [x, xg2])

#define USE_ASYNC_LDS 1

static __device__ __forceinline__ v8f wmma_f16(v16h a, v16h b, v8f c) {
  // 8 args: (neg_a, A, neg_b, B, c_mod, C, reuse_a, reuse_b)
  return __builtin_amdgcn_wmma_f32_16x16x32_f16(false, a, false, b, (short)0, c,
                                                false, false);
}

// A-operand (16x32 f16) per-lane K index for VGPR-half pair v (ISA 7.12.2):
// lanes 0-15: V0..3 -> K{0..7}, V4..7 -> K{16..23}; lanes 16-31: +8
static __device__ __forceinline__ int a_k0(int v, int hi) {
  return 2 * v + (v >= 4 ? 8 : 0) + 8 * hi;
}

// ---- CDNA5 async global->LDS (ASYNCcnt-tracked DMA, no VGPR round trip) ----
static __device__ __forceinline__ void async_b128(const _Float16* g, _Float16* l) {
#if USE_ASYNC_LDS
  // Generic address of a __shared__ object carries the LDS byte offset in
  // bits [31:0] (ISA 10.2 aperture mapping), which is what VDST expects.
  unsigned int loff = (unsigned int)(unsigned long long)(void*)l;
  asm volatile("global_load_async_to_lds_b128 %0, %1, off"
               :: "v"(loff), "v"(g) : "memory");
#else
  *(uint4*)l = *(const uint4*)g;
#endif
}
static __device__ __forceinline__ void wait_async() {
#if USE_ASYNC_LDS
  asm volatile("s_wait_asynccnt 0x0" ::: "memory");
#endif
}

// ---------------------------------------------------------------------------
// prep: ias_f16[bt][n][96] = concat(x[bt,n,:32], states[b, 10+t, n, :64])
// ---------------------------------------------------------------------------
__global__ void k_ias(const float* __restrict__ x, const float* __restrict__ states,
                      _Float16* __restrict__ ias) {
  size_t idx = (size_t)blockIdx.x * blockDim.x + threadIdx.x;
  if (idx >= (size_t)BT * NN * CIN) return;
  int e = (int)(idx % CIN);
  size_t nt = idx / CIN;
  int n = (int)(nt % NN);
  int bt = (int)(nt / NN);
  int b = bt >> 1, t = bt & 1;
  float v;
  if (e < 32) v = x[((size_t)bt * NN + n) * 32 + e];
  else        v = states[((size_t)(b * TT + 10 + t) * NN + n) * 64 + (e - 32)];
  ias[idx] = (_Float16)v;
}

// cand_f16[bt][n][96] = concat(x, z * state)
__global__ void k_cand(const float* __restrict__ x, const float* __restrict__ states,
                       const float* __restrict__ zbuf, _Float16* __restrict__ cand) {
  size_t idx = (size_t)blockIdx.x * blockDim.x + threadIdx.x;
  if (idx >= (size_t)BT * NN * CIN) return;
  int e = (int)(idx % CIN);
  size_t nt = idx / CIN;
  int n = (int)(nt % NN);
  int bt = (int)(nt / NN);
  int b = bt >> 1, t = bt & 1;
  float v;
  if (e < 32) v = x[((size_t)bt * NN + n) * 32 + e];
  else {
    int o = e - 32;
    float st = states[((size_t)(b * TT + 10 + t) * NN + n) * 64 + o];
    float zv = zbuf[((size_t)bt * NN + n) * 64 + o];
    v = zv * st;
  }
  cand[idx] = (_Float16)v;
}

// ---------------------------------------------------------------------------
// emb = LN(node_emb + time_emb) over DE=16, eps=1e-12; zero-pad d to 32.
// embQ layout [bt][n][32] (A-operand source), embT layout [bt][d=32][n] (B src)
// ---------------------------------------------------------------------------
__global__ void k_emb(const float* __restrict__ node_emb, const float* __restrict__ time_emb,
                      const float* __restrict__ gw, const float* __restrict__ gb,
                      _Float16* __restrict__ embQ, _Float16* __restrict__ embT) {
  int idx = blockIdx.x * blockDim.x + threadIdx.x;
  if (idx >= BT * NN) return;
  int n = idx % NN, bt = idx / NN;
  float e[16];
  float mean = 0.f;
#pragma unroll
  for (int d = 0; d < 16; ++d) { e[d] = node_emb[n * 16 + d] + time_emb[bt * 16 + d]; mean += e[d]; }
  mean *= (1.f / 16.f);
  float var = 0.f;
#pragma unroll
  for (int d = 0; d < 16; ++d) { float dd = e[d] - mean; var += dd * dd; }
  var *= (1.f / 16.f);
  float rs = rsqrtf(var + 1e-12f);
#pragma unroll
  for (int d = 0; d < 16; ++d) {
    float v = (e[d] - mean) * rs * gw[d] + gb[d];
    embQ[((size_t)bt * NN + n) * 32 + d] = (_Float16)v;
    embT[((size_t)bt * 32 + d) * NN + n] = (_Float16)v;
  }
#pragma unroll
  for (int d = 16; d < 32; ++d) {
    embQ[((size_t)bt * NN + n) * 32 + d] = (_Float16)0.f;
    embT[((size_t)bt * 32 + d) * NN + n] = (_Float16)0.f;
  }
}

// W_g[n][ki=192][o=64] = sum_d node_emb[n,d] * Wp[d][ki][o]   (f16 out)
__global__ void k_W(const float* __restrict__ node_emb, const float* __restrict__ Wp,
                    _Float16* __restrict__ Wg) {
  size_t idx = (size_t)blockIdx.x * blockDim.x + threadIdx.x;
  if (idx >= (size_t)NN * (KI * 64)) return;
  int n = (int)(idx / (KI * 64));
  int kio = (int)(idx % (KI * 64));
  float s = 0.f;
#pragma unroll
  for (int d = 0; d < 16; ++d) s += node_emb[n * 16 + d] * Wp[(size_t)d * (KI * 64) + kio];
  Wg[idx] = (_Float16)s;
}

// bias[bt][o] = sum_d time_emb[bt,d] * bp[d,o]
__global__ void k_bias(const float* __restrict__ time_emb, const float* __restrict__ bp,
                       float* __restrict__ biasg) {
  int idx = blockIdx.x * blockDim.x + threadIdx.x;
  if (idx >= BT * 64) return;
  int o = idx % 64, bt = idx / 64;
  float s = 0.f;
#pragma unroll
  for (int d = 0; d < 16; ++d) s += time_emb[bt * 16 + d] * bp[d * 64 + o];
  biasg[idx] = s;
}

// ---------------------------------------------------------------------------
// Stage 1: fused  xg2[bt, q, :96] = softmax_m( embQ . embK^T ) @ xsrc
// Flash-style online softmax. grid = (N/128, BT), block = 256 (8 waves).
// Each wave owns 16 query rows. Key loop in blocks of 128, double-buffered
// in LDS via CDNA5 async global->LDS DMA (ASYNCcnt).
// ---------------------------------------------------------------------------

// Issue one chunk's DMA: embT chunk [32 d][128 key] (8 KB) + X chunk
// [128 key][96 c] (24 KB). 256 threads, 16 B per transfer, all aligned.
static __device__ __forceinline__ void issue_chunk(const _Float16* __restrict__ embT,
                                                   const _Float16* __restrict__ xsrc,
                                                   int bt, int kb,
                                                   _Float16* dET, _Float16* dX, int tid) {
#pragma unroll
  for (int j = 0; j < 2; ++j) {          // 512 x 16B for embT chunk
    int i = tid + 256 * j;
    int d = i >> 4, seg = i & 15;
    const _Float16* g = embT + ((size_t)bt * 32 + d) * NN + kb * 128 + seg * 8;
    async_b128(g, dET + d * 128 + seg * 8);
  }
#pragma unroll
  for (int j = 0; j < 6; ++j) {          // 1536 x 16B for X chunk
    int i = tid + 256 * j;
    int r = i / 12, seg = i - r * 12;
    const _Float16* g = xsrc + ((size_t)bt * NN + kb * 128 + r) * CIN + seg * 8;
    async_b128(g, dX + r * CIN + seg * 8);
  }
}

__global__ __launch_bounds__(256)
void k_s1(const _Float16* __restrict__ embQ, const _Float16* __restrict__ embT,
          const _Float16* __restrict__ xsrc, _Float16* __restrict__ xg2) {
  __shared__ __align__(16) _Float16 sETd[2][32 * 128];    // 2 x 8 KB
  __shared__ __align__(16) _Float16 sXd[2][128 * CIN];    // 2 x 24 KB
  __shared__ __align__(16) _Float16 sP[8][16 * 128];      // per-wave P, 32 KB

  const int bt   = blockIdx.y;
  const int tid  = threadIdx.x;
  const int wave = tid >> 5;
  const int lane = tid & 31;
  const int row  = lane & 15;
  const int hi   = lane >> 4;
  const int qbase = blockIdx.x * 128 + wave * 16;

  // ---- Q fragment (A operand 16x32; d 16..31 zero-padded) ----
  v16h qf;
  {
    const _Float16* qp = embQ + ((size_t)bt * NN + (qbase + row)) * 32;
#pragma unroll
    for (int v = 0; v < 8; ++v) {
      int k0 = a_k0(v, hi);
      qf[2 * v]     = qp[k0];
      qf[2 * v + 1] = qp[k0 + 1];
    }
  }

  v8f oacc[6];
#pragma unroll
  for (int c = 0; c < 6; ++c) oacc[c] = (v8f){0.f, 0.f, 0.f, 0.f, 0.f, 0.f, 0.f, 0.f};
  float mrow[8], lrow[8];
#pragma unroll
  for (int j = 0; j < 8; ++j) { mrow[j] = -1e30f; lrow[j] = 0.f; }

  // prologue: DMA chunk 0 into buffer 0
  issue_chunk(embT, xsrc, bt, 0, sETd[0], sXd[0], tid);

  for (int kb = 0; kb < NN / 128; ++kb) {
    const int cur = kb & 1;
    wait_async();        // this wave's outstanding chunk DMA complete
    __syncthreads();     // -> every wave's share complete; prev buffer free
    if (kb + 1 < NN / 128)
      issue_chunk(embT, xsrc, bt, kb + 1, sETd[cur ^ 1], sXd[cur ^ 1], tid);

    const _Float16* sET = sETd[cur];
    const _Float16* sX  = sXd[cur];

    // ---- S = Q K^T : 8 tiles of 16 keys (K dim = d, zero-padded 32) ----
    v8f st[8];
#pragma unroll
    for (int j = 0; j < 8; ++j) {
      v16h bf;
      // B operand: lane indexes N (=key), elements index K (=d) in A's k order
#pragma unroll
      for (int v = 0; v < 8; ++v) {
        int kd = a_k0(v, hi);
        bf[2 * v]     = sET[kd * 128 + j * 16 + row];
        bf[2 * v + 1] = sET[(kd + 1) * 128 + j * 16 + row];
      }
      v8f z = (v8f){0.f, 0.f, 0.f, 0.f, 0.f, 0.f, 0.f, 0.f};
      st[j] = wmma_f16(qf, bf, z);
    }

    // ---- online softmax stats: rows live at VGPR j (+8 for high half) ----
    float bm[8];
#pragma unroll
    for (int j = 0; j < 8; ++j) bm[j] = -1e30f;
#pragma unroll
    for (int t8 = 0; t8 < 8; ++t8)
#pragma unroll
      for (int j = 0; j < 8; ++j) bm[j] = fmaxf(bm[j], st[t8][j]);
#pragma unroll
    for (int j = 0; j < 8; ++j)
#pragma unroll
      for (int off = 1; off < 16; off <<= 1)
        bm[j] = fmaxf(bm[j], __shfl_xor(bm[j], off, 16));

    float scl[8];
#pragma unroll
    for (int j = 0; j < 8; ++j) {
      float nm = fmaxf(mrow[j], bm[j]);
      scl[j] = __expf(mrow[j] - nm);
      mrow[j] = nm;
      lrow[j] *= scl[j];
    }
#pragma unroll
    for (int c = 0; c < 6; ++c)
#pragma unroll
      for (int j = 0; j < 8; ++j) oacc[c][j] *= scl[j];

    float ps[8];
#pragma unroll
    for (int j = 0; j < 8; ++j) ps[j] = 0.f;
#pragma unroll
    for (int t8 = 0; t8 < 8; ++t8) {
#pragma unroll
      for (int j = 0; j < 8; ++j) {
        float p = __expf(st[t8][j] - mrow[j]);
        ps[j] += p;
        sP[wave][(j + 8 * hi) * 128 + t8 * 16 + row] = (_Float16)p;
      }
    }
#pragma unroll
    for (int j = 0; j < 8; ++j) {
#pragma unroll
      for (int off = 1; off < 16; off <<= 1) ps[j] += __shfl_xor(ps[j], off, 16);
      lrow[j] += ps[j];
    }

    // ---- O += P @ X : 4 k-steps of 32 keys x 6 output c-tiles ----
#pragma unroll
    for (int kk = 0; kk < 4; ++kk) {
      v16h af;
#pragma unroll
      for (int v = 0; v < 8; ++v) {
        int k0 = a_k0(v, hi) + kk * 32;
        af[2 * v]     = sP[wave][row * 128 + k0];
        af[2 * v + 1] = sP[wave][row * 128 + k0 + 1];
      }
#pragma unroll
      for (int c = 0; c < 6; ++c) {
        v16h bf;
#pragma unroll
        for (int v = 0; v < 8; ++v) {
          int kkey = a_k0(v, hi) + kk * 32;
          bf[2 * v]     = sX[kkey * CIN + c * 16 + row];
          bf[2 * v + 1] = sX[(kkey + 1) * CIN + c * 16 + row];
        }
        oacc[c] = wmma_f16(af, bf, oacc[c]);
      }
    }
    // next iteration's top barrier protects buffer reuse
  }

  // ---- finalize: divide by row sum, store f16 ----
  float inv[8];
#pragma unroll
  for (int j = 0; j < 8; ++j) inv[j] = (lrow[j] > 0.f) ? 1.f / lrow[j] : 0.f;
#pragma unroll
  for (int c = 0; c < 6; ++c)
#pragma unroll
    for (int j = 0; j < 8; ++j) {
      size_t q = (size_t)qbase + j + 8 * hi;
      xg2[((size_t)bt * NN + q) * CIN + c * 16 + row] = (_Float16)(oacc[c][j] * inv[j]);
    }
}

// ---------------------------------------------------------------------------
// Stage 2: per-node GEMM (32x192 @ 192x64) + bias + LN + 4-head attention
// over 12 keys + gate activation / GRU combine.
// grid = N/4, block = 128 (one node per wave).
// mode 0: out = sigmoid(tag)                (z / r gates)
// mode 2: out = r*state + (1-r)*tanh(tag)   (final output)
// ---------------------------------------------------------------------------
__global__ __launch_bounds__(128)
void k_s2(const _Float16* __restrict__ xin, const _Float16* __restrict__ xg2,
          const _Float16* __restrict__ Wg, const float* __restrict__ biasg,
          const float* __restrict__ anw, const float* __restrict__ anb,
          const float* __restrict__ states, const float* __restrict__ rbuf,
          float* __restrict__ outp, int mode) {
  __shared__ float sG[4][32 * 64];  // 32 KB
  const int wave = threadIdx.x >> 5;
  const int lane = threadIdx.x & 31;
  const int row  = lane & 15;
  const int hi   = lane >> 4;
  const int n = blockIdx.x * 4 + wave;

  v8f acc[2][4];
#pragma unroll
  for (int m = 0; m < 2; ++m)
#pragma unroll
    for (int c = 0; c < 4; ++c) acc[m][c] = (v8f){0.f, 0.f, 0.f, 0.f, 0.f, 0.f, 0.f, 0.f};

#pragma unroll
  for (int kk = 0; kk < 6; ++kk) {
    v16h af[2];
#pragma unroll
    for (int m = 0; m < 2; ++m) {
      int btr = m * 16 + row;
#pragma unroll
      for (int v = 0; v < 8; ++v) {
        int k = kk * 32 + a_k0(v, hi);
        const _Float16* sp = (k < CIN)
            ? xin + ((size_t)btr * NN + n) * CIN + k
            : xg2 + ((size_t)btr * NN + n) * CIN + (k - CIN);
        af[m][2 * v]     = sp[0];
        af[m][2 * v + 1] = sp[1];
      }
    }
#pragma unroll
    for (int c = 0; c < 4; ++c) {
      v16h bf;
#pragma unroll
      for (int v = 0; v < 8; ++v) {
        int k = kk * 32 + a_k0(v, hi);  // B operand K index, same order as A
        const _Float16* wp = Wg + ((size_t)n * KI + k) * 64 + c * 16 + row;
        bf[2 * v]     = wp[0];
        bf[2 * v + 1] = wp[64];
      }
#pragma unroll
      for (int m = 0; m < 2; ++m) acc[m][c] = wmma_f16(af[m], bf, acc[m][c]);
    }
  }

  // D tiles + bias -> LDS (rows = bt, cols = o)
#pragma unroll
  for (int m = 0; m < 2; ++m)
#pragma unroll
    for (int c = 0; c < 4; ++c)
#pragma unroll
      for (int j = 0; j < 8; ++j) {
        int btr = m * 16 + j + 8 * hi;
        int o = c * 16 + row;
        sG[wave][btr * 64 + o] = acc[m][c][j] + biasg[btr * 64 + o];
      }
  __syncthreads();

  // epilogue: one (b,t) row per lane
  {
    int bt = lane;
    int b = bt >> 1, t = bt & 1;
    float* grow = &sG[wave][bt * 64];
    float mean = 0.f;
#pragma unroll
    for (int o = 0; o < 64; ++o) mean += grow[o];
    mean *= (1.f / 64.f);
    float var = 0.f;
#pragma unroll
    for (int o = 0; o < 64; ++o) { float d = grow[o] - mean; var += d * d; }
    var *= (1.f / 64.f);
    float rs = rsqrtf(var + 1e-5f);

#pragma unroll
    for (int h = 0; h < 4; ++h) {
      float qh[16];
#pragma unroll
      for (int d = 0; d < 16; ++d) {
        int o = h * 16 + d;
        qh[d] = (grow[o] - mean) * rs * anw[o] + anb[o];
      }
      float sc[TT];
      float mx = -1e30f;
      for (int tk = 0; tk < TT; ++tk) {
        const float* kv = states + ((size_t)(b * TT + tk) * NN + n) * 64 + h * 16;
        float s = 0.f;
#pragma unroll
        for (int d = 0; d < 16; ++d) s += qh[d] * kv[d];
        s *= 0.25f;  // 1/sqrt(hd=16)
        sc[tk] = s;
        mx = fmaxf(mx, s);
      }
      float se = 0.f;
      for (int tk = 0; tk < TT; ++tk) { sc[tk] = __expf(sc[tk] - mx); se += sc[tk]; }
      float isum = 1.f / se;
      float ah[16];
#pragma unroll
      for (int d = 0; d < 16; ++d) ah[d] = 0.f;
      for (int tk = 0; tk < TT; ++tk) {
        const float* kv = states + ((size_t)(b * TT + tk) * NN + n) * 64 + h * 16;
        float p = sc[tk] * isum;
#pragma unroll
        for (int d = 0; d < 16; ++d) ah[d] += p * kv[d];
      }
#pragma unroll
      for (int d = 0; d < 16; ++d) {
        int o = h * 16 + d;
        float tag = grow[o] + ah[d];
        size_t idx = ((size_t)bt * NN + n) * 64 + o;
        if (mode == 0) {
          outp[idx] = 1.f / (1.f + __expf(-tag));
        } else {
          float hc = tanhf(tag);
          float rv = rbuf[idx];
          float stv = states[((size_t)(b * TT + 10 + t) * NN + n) * 64 + o];
          outp[idx] = rv * stv + (1.f - rv) * hc;
        }
      }
    }
  }
}

// ---------------------------------------------------------------------------
// Host launch
// ---------------------------------------------------------------------------
extern "C" void kernel_launch(void* const* d_in, const int* in_sizes, int n_in,
                              void* d_out, int out_size, void* d_ws, size_t ws_size,
                              hipStream_t stream) {
  (void)in_sizes; (void)n_in; (void)out_size; (void)ws_size;
  const float* x        = (const float*)d_in[0];
  const float* states   = (const float*)d_in[1];
  const float* node_emb = (const float*)d_in[2];
  const float* time_emb = (const float*)d_in[3];
  // per gate g: Wp=4+6g, bp=5+6g, gnw=6+6g, gnb=7+6g, anw=8+6g, anb=9+6g
  const float* Wp[3]  = {(const float*)d_in[4],  (const float*)d_in[10], (const float*)d_in[16]};
  const float* bp[3]  = {(const float*)d_in[5],  (const float*)d_in[11], (const float*)d_in[17]};
  const float* gnw[3] = {(const float*)d_in[6],  (const float*)d_in[12], (const float*)d_in[18]};
  const float* gnb[3] = {(const float*)d_in[7],  (const float*)d_in[13], (const float*)d_in[19]};
  const float* anw[3] = {(const float*)d_in[8],  (const float*)d_in[14], (const float*)d_in[20]};
  const float* anb[3] = {(const float*)d_in[9],  (const float*)d_in[15], (const float*)d_in[21]};

  // workspace layout (bytes)
  char* ws = (char*)d_ws;
  size_t off = 0;
  auto alloc = [&](size_t bytes) { char* p = ws + off; off = (off + bytes + 255) & ~(size_t)255; return p; };
  const size_t elems_x96 = (size_t)BT * NN * CIN;              // 6.29 M
  _Float16* ias  = (_Float16*)alloc(elems_x96 * 2);            // 12.6 MB
  _Float16* cand = (_Float16*)alloc(elems_x96 * 2);            // 12.6 MB
  _Float16* xg2  = (_Float16*)alloc(elems_x96 * 2);            // 12.6 MB (reused per gate)
  _Float16* Wbuf = (_Float16*)alloc((size_t)NN * KI * 64 * 2); // 50.3 MB (reused per gate)
  _Float16* embQ[3];
  _Float16* embT[3];
  for (int g = 0; g < 3; ++g) {
    embQ[g] = (_Float16*)alloc((size_t)BT * NN * 32 * 2);      // 4.2 MB each
    embT[g] = (_Float16*)alloc((size_t)BT * NN * 32 * 2);
  }
  float* biasg[3];
  for (int g = 0; g < 3; ++g) biasg[g] = (float*)alloc((size_t)BT * 64 * 4);
  float* zbuf = (float*)alloc((size_t)BT * NN * 64 * 4);       // 16.8 MB
  float* rbuf = (float*)alloc((size_t)BT * NN * 64 * 4);       // 16.8 MB
  float* outp = (float*)d_out;

  // --- prep ---
  {
    size_t tot = elems_x96;
    k_ias<<<dim3((tot + 255) / 256), dim3(256), 0, stream>>>(x, states, ias);
  }
  for (int g = 0; g < 3; ++g) {
    k_emb<<<dim3((BT * NN + 255) / 256), dim3(256), 0, stream>>>(
        node_emb, time_emb, gnw[g], gnb[g], embQ[g], embT[g]);
    k_bias<<<dim3((BT * 64 + 255) / 256), dim3(256), 0, stream>>>(time_emb, bp[g], biasg[g]);
  }

  dim3 g1(NN / 128, BT), b1(256);
  dim3 g2(NN / 4), b2(128);
  size_t wtot = (size_t)NN * KI * 64;

  // --- gate z ---
  k_s1<<<g1, b1, 0, stream>>>(embQ[0], embT[0], ias, xg2);
  k_W<<<dim3((wtot + 255) / 256), dim3(256), 0, stream>>>(node_emb, Wp[0], Wbuf);
  k_s2<<<g2, b2, 0, stream>>>(ias, xg2, Wbuf, biasg[0], anw[0], anb[0], states, rbuf, zbuf, 0);

  // --- gate r ---
  k_s1<<<g1, b1, 0, stream>>>(embQ[1], embT[1], ias, xg2);
  k_W<<<dim3((wtot + 255) / 256), dim3(256), 0, stream>>>(node_emb, Wp[1], Wbuf);
  k_s2<<<g2, b2, 0, stream>>>(ias, xg2, Wbuf, biasg[1], anw[1], anb[1], states, rbuf, rbuf, 0);

  // --- candidate input ---
  k_cand<<<dim3((elems_x96 + 255) / 256), dim3(256), 0, stream>>>(x, states, zbuf, cand);

  // --- gate u + final combine ---
  k_s1<<<g1, b1, 0, stream>>>(embQ[2], embT[2], cand, xg2);
  k_W<<<dim3((wtot + 255) / 256), dim3(256), 0, stream>>>(node_emb, Wp[2], Wbuf);
  k_s2<<<g2, b2, 0, stream>>>(cand, xg2, Wbuf, biasg[2], anw[2], anb[2], states, rbuf, outp, 2);
}
